// PatchcoreModel_86973087744664
// MI455X (gfx1250) — matrix operand order repaired and checked by reading, not compiled, not run on hardware
//
#include <hip/hip_runtime.h>

// ---------------------------------------------------------------------------
// PatchCore scoring on MI455X (gfx1250):
//   d2 = ||x||^2 + ||m||^2 - 2 x.m ;  scores = sqrt(max(d2,0)) ; top-9 smallest
// Compute-bound (AI ~2700 FLOP/B) -> bf16 WMMA, fp32 accumulation.
// bf16 bank (123 MB) is L2-resident (192 MB global L2); register blocking
// targets 1.5 global_load_b128 per v_wmma (was 3.0 in v1).
// ---------------------------------------------------------------------------

#define N_QUERY 6272
#define D_EMBED 1536
#define M_BANK  40000
#define K_NN    9

#define ROWS_PER_BLOCK 64          // 4 row subtiles of 16
#define CHUNK_COLS     256         // 8 waves x 2 col subtiles of 16
#define COL_SPLIT      4
#define COLS_PER_SPLIT (M_BANK / COL_SPLIT)   // 10000
#define THREADS        256
#define TOP9_INF       3.0e38f

typedef __attribute__((ext_vector_type(16))) __bf16 v16bf;
typedef __attribute__((ext_vector_type(8)))  float  v8f;

struct Bytes32 { uint4 lo, hi; };

__device__ __forceinline__ v16bf make_frag(uint4 lo, uint4 hi) {
  Bytes32 b{lo, hi};
  return __builtin_bit_cast(v16bf, b);
}

__device__ __forceinline__ unsigned short f32_to_bf16_rne(float f) {
  unsigned int u = __builtin_bit_cast(unsigned int, f);
  u += 0x7FFFu + ((u >> 16) & 1u);          // round-to-nearest-even
  return (unsigned short)(u >> 16);
}

// Sorted ascending top-9 kept entirely in registers: guarded unrolled bubble.
__device__ __forceinline__ void top9_insert(float t[K_NN], float v) {
  if (v < t[8]) {
    t[8] = v;
#pragma unroll
    for (int i = 8; i > 0; --i) {
      float a = t[i - 1], b = t[i];
      t[i - 1] = fminf(a, b);
      t[i]     = fmaxf(a, b);
    }
  }
}

// ---- kernel 1: fp32 -> bf16 conversion + row squared norms ----------------
__global__ void __launch_bounds__(128)
conv_norm_kernel(const float* __restrict__ src,
                 unsigned short* __restrict__ dst,
                 float* __restrict__ nrm) {
  const int row = blockIdx.x;
  const float* s = src + (size_t)row * D_EMBED;
  unsigned short* d = dst + (size_t)row * D_EMBED;
  float acc = 0.f;
  for (int i = threadIdx.x; i < D_EMBED; i += 128) {
    float v = s[i];
    acc = fmaf(v, v, acc);
    d[i] = f32_to_bf16_rne(v);
  }
  __shared__ float red[128];
  red[threadIdx.x] = acc;
  __syncthreads();
  for (int st = 64; st > 0; st >>= 1) {
    if (threadIdx.x < st) red[threadIdx.x] += red[threadIdx.x + st];
    __syncthreads();
  }
  if (threadIdx.x == 0) nrm[row] = red[0];
}

// ---- kernel 2: fused bf16-WMMA GEMM + per-row top-9 -----------------------
// Block: 256 threads (8 waves), 64 query rows, one column-split quarter.
// Wave w owns columns [32w, 32w+32) of each 256-col chunk: 2 col subtiles
// x 4 row subtiles = 8 independent 16x16 f32 accumulators.
// Out-of-range columns are clamped for addressing (uniform WMMA, no
// divergence) and excluded exactly in the top-k merge phase.
__global__ void __launch_bounds__(THREADS)
patchcore_topk_kernel(const unsigned short* __restrict__ Xb,
                      const unsigned short* __restrict__ Mb,
                      const float* __restrict__ x2,
                      const float* __restrict__ m2,
                      float* __restrict__ part) {
  __shared__ float tile[ROWS_PER_BLOCK][CHUNK_COLS + 4];   // +4 pad: banks
  __shared__ float ktop[ROWS_PER_BLOCK][4][K_NN];

  const int tid  = threadIdx.x;
  const int wave = tid >> 5;
  const int lane = tid & 31;
  const int g    = lane >> 4;   // lane group (ISA 16-bit A/B layouts)
  const int ln   = lane & 15;

  const int rowBase = blockIdx.x * ROWS_PER_BLOCK;
  const int colBase = blockIdx.y * COLS_PER_SPLIT;

  // merge-phase view: thread t scans row (t&63), cols [64*(t>>6), +64)
  const int mrow   = tid & 63;
  const int mslice = tid >> 6;

  float t9[K_NN];
#pragma unroll
  for (int i = 0; i < K_NN; ++i) t9[i] = TOP9_INF;

  // A fragment bases: row m = ln of each 16-row subtile, contiguous in K.
  const unsigned short* Ap[4];
#pragma unroll
  for (int s = 0; s < 4; ++s)
    Ap[s] = Xb + (size_t)(rowBase + 16 * s + ln) * D_EMBED;

  for (int cb = 0; cb < COLS_PER_SPLIT; cb += CHUNK_COLS) {
    const int rem   = COLS_PER_SPLIT - cb;
    const int ccols = (rem < CHUNK_COLS) ? rem : CHUNK_COLS;

    // bank rows (= B columns) for this wave's two column subtiles, clamped
    const int col0 = colBase + cb + wave * 32 + ln;
    const int col1 = col0 + 16;
    const int col0c = (col0 < M_BANK) ? col0 : (M_BANK - 1);
    const int col1c = (col1 < M_BANK) ? col1 : (M_BANK - 1);
    const unsigned short* B0 = Mb + (size_t)col0c * D_EMBED;
    const unsigned short* B1 = Mb + (size_t)col1c * D_EMBED;
    __builtin_prefetch(B0 + (size_t)CHUNK_COLS * D_EMBED, 0, 1); // next chunk

    v8f acc[8];
#pragma unroll
    for (int i = 0; i < 8; ++i) acc[i] = (v8f){};

    for (int k0 = 0; k0 < D_EMBED; k0 += 32) {
      // A 16x32 bf16: lanes 0-15 K 0..15, lanes 16-31 K 16..31 (contig 32B)
      const int ka = k0 + 16 * g;
      // B 32x16 bf16: lanes 0-15 K 0..15, lanes 16-31 K 16..31 (contig 32B)
      const int kb = k0 + 16 * g;

      uint4 b0lo = *(const uint4*)(B0 + kb);
      uint4 b0hi = *(const uint4*)(B0 + kb + 8);
      uint4 b1lo = *(const uint4*)(B1 + kb);
      uint4 b1hi = *(const uint4*)(B1 + kb + 8);
      v16bf bf0 = make_frag(b0lo, b0hi);
      v16bf bf1 = make_frag(b1lo, b1hi);

#pragma unroll
      for (int s = 0; s < 4; ++s) {
        uint4 alo = *(const uint4*)(Ap[s] + ka);
        uint4 ahi = *(const uint4*)(Ap[s] + ka + 8);
        v16bf af = make_frag(alo, ahi);
        acc[2 * s] = __builtin_amdgcn_wmma_f32_16x16x32_bf16(
            false, af, false, bf0, (short)0, acc[2 * s], false, false);
        acc[2 * s + 1] = __builtin_amdgcn_wmma_f32_16x16x32_bf16(
            false, af, false, bf1, (short)0, acc[2 * s + 1], false, false);
      }
    }

    // s = ||m||^2 - 2*dot  (||x||^2 row-constant -> deferred; monotonic)
    const float mm0 = m2[col0c];
    const float mm1 = m2[col1c];
    const int wc = wave * 32 + ln;
#pragma unroll
    for (int s = 0; s < 4; ++s) {
#pragma unroll
      for (int r = 0; r < 8; ++r) {
        tile[16 * s + 8 * g + r][wc]      = fmaf(-2.f, acc[2 * s][r], mm0);
        tile[16 * s + 8 * g + r][wc + 16] = fmaf(-2.f, acc[2 * s + 1][r], mm1);
      }
    }
    __syncthreads();

#pragma unroll 4
    for (int j = 0; j < 64; ++j) {
      const int c = mslice * 64 + j;
      if (c < ccols) top9_insert(t9, tile[mrow][c]);
    }
    __syncthreads();
  }

  // spill per-thread partial top-9, then one thread per row merges 4 slices
#pragma unroll
  for (int i = 0; i < K_NN; ++i) ktop[mrow][mslice][i] = t9[i];
  __syncthreads();

  if (tid < ROWS_PER_BLOCK) {
    float f9[K_NN];
#pragma unroll
    for (int i = 0; i < K_NN; ++i) f9[i] = TOP9_INF;
    for (int s = 0; s < 4; ++s) {
#pragma unroll
      for (int i = 0; i < K_NN; ++i) top9_insert(f9, ktop[tid][s][i]);
    }
    const float xx = x2[rowBase + tid];
    float* p = part + ((size_t)(rowBase + tid) * COL_SPLIT + blockIdx.y) * K_NN;
#pragma unroll
    for (int i = 0; i < K_NN; ++i)
      p[i] = sqrtf(fmaxf(xx + f9[i], 0.f));   // monotonic -> order preserved
  }
}

// ---- kernel 3: merge the COL_SPLIT sorted partials per row ----------------
__global__ void __launch_bounds__(256)
final_merge_kernel(const float* __restrict__ part, float* __restrict__ out) {
  const int row = blockIdx.x * blockDim.x + threadIdx.x;
  if (row >= N_QUERY) return;
  const float* p = part + (size_t)row * COL_SPLIT * K_NN;
  float t9[K_NN];
#pragma unroll
  for (int i = 0; i < K_NN; ++i) t9[i] = TOP9_INF;
  for (int i = 0; i < COL_SPLIT * K_NN; ++i) top9_insert(t9, p[i]);
  float* o = out + (size_t)row * K_NN;
#pragma unroll
  for (int k = 0; k < K_NN; ++k) o[k] = t9[k];
}

// ---------------------------------------------------------------------------
extern "C" void kernel_launch(void* const* d_in, const int* in_sizes, int n_in,
                              void* d_out, int out_size, void* d_ws, size_t ws_size,
                              hipStream_t stream) {
  (void)in_sizes; (void)n_in; (void)out_size; (void)ws_size;
  const float* emb  = (const float*)d_in[0];   // [N_QUERY, D_EMBED] fp32
  const float* bank = (const float*)d_in[1];   // [M_BANK,  D_EMBED] fp32
  // d_in[2] = n_neighbors scalar (fixed K_NN = 9)

  char* ws = (char*)d_ws;
  size_t off = 0;
  auto take = [&](size_t bytes) -> void* {
    void* p = ws + off;
    off += (bytes + 255) & ~(size_t)255;
    return p;
  };
  unsigned short* Xb   = (unsigned short*)take((size_t)N_QUERY * D_EMBED * 2);
  unsigned short* Mb   = (unsigned short*)take((size_t)M_BANK  * D_EMBED * 2);
  float*          x2   = (float*)take((size_t)N_QUERY * sizeof(float));
  float*          m2   = (float*)take((size_t)M_BANK  * sizeof(float));
  float*          part = (float*)take((size_t)N_QUERY * COL_SPLIT * K_NN * sizeof(float));

  conv_norm_kernel<<<N_QUERY, 128, 0, stream>>>(emb,  Xb, x2);
  conv_norm_kernel<<<M_BANK,  128, 0, stream>>>(bank, Mb, m2);

  dim3 grid(N_QUERY / ROWS_PER_BLOCK, COL_SPLIT);
  patchcore_topk_kernel<<<grid, THREADS, 0, stream>>>(Xb, Mb, x2, m2, part);

  final_merge_kernel<<<(N_QUERY + 255) / 256, 256, 0, stream>>>(part, (float*)d_out);
}
